// CrossAttentionBlock_30958124269668
// MI455X (gfx1250) — matrix operand order repaired and checked by reading, not compiled, not run on hardware
//
#include <hip/hip_runtime.h>
#include <hip/hip_bf16.h>

typedef __attribute__((ext_vector_type(16))) _Float16 v16h;
typedef __attribute__((ext_vector_type(8)))  _Float16 v8h;
typedef __attribute__((ext_vector_type(4)))  _Float16 v4h;
typedef __attribute__((ext_vector_type(2)))  _Float16 v2h;
typedef __attribute__((ext_vector_type(8)))  float    v8f;
typedef __attribute__((ext_vector_type(4)))  unsigned int u32x4;
typedef __attribute__((ext_vector_type(8)))  int          i32x8;
typedef __attribute__((ext_vector_type(4)))  int          i32x4;

#define BATCH 4
#define CCH   512
#define LPIX  4096
#define CCC   64
#define LCTX  1024
#define NHEAD 8
#define HD    64

// ---------------- helpers ----------------
__device__ inline v8f v8f_zero() {
  v8f z;
#pragma unroll
  for (int i = 0; i < 8; ++i) z[i] = 0.f;
  return z;
}

__device__ inline v16h cat8(v8h lo, v8h hh) {
  v16h r;
#pragma unroll
  for (int i = 0; i < 8; ++i) { r[i] = lo[i]; r[8 + i] = hh[i]; }
  return r;
}

// A-fragment (16x32 f16): p = row base (+k-chunk base). Lane half `hi` owns
// K = {hi*16 .. hi*16+15}.
__device__ inline v16h frag_a(const _Float16* p, int hi) {
  v8h lo = *(const v8h*)(p + hi * 16);
  v8h hh = *(const v8h*)(p + hi * 16 + 8);
  return cat8(lo, hh);
}

// B-fragment (32x16 f16): p points at 16 contiguous K values for this lane.
__device__ inline v16h frag_b(const _Float16* p) {
  v8h lo = *(const v8h*)(p);
  v8h hh = *(const v8h*)(p + 8);
  return cat8(lo, hh);
}

__device__ inline v8f wmma_f16(v16h a, v16h b, v8f c) {
  return __builtin_amdgcn_wmma_f32_16x16x32_f16(false, a, false, b,
                                                (short)0, c, false, false);
}

// ---- DPP16 cross-lane reductions within a 16-lane row (VALU, no LDS) ----
template <int CTRL>
__device__ inline float dpp_movf(float v) {
  return __int_as_float(__builtin_amdgcn_update_dpp(
      0, __float_as_int(v), CTRL, 0xF, 0xF, true));
}

__device__ inline float row16_max(float v) {
  v = fmaxf(v, dpp_movf<0xB1>(v));   // xor1 (quad_perm 1,0,3,2)
  v = fmaxf(v, dpp_movf<0x4E>(v));   // xor2 (quad_perm 2,3,0,1)
  v = fmaxf(v, dpp_movf<0x141>(v));  // xor4-equiv (row_half_mirror)
  v = fmaxf(v, dpp_movf<0x140>(v));  // xor8-equiv (row_mirror)
  return v;
}

__device__ inline float row16_sum(float v) {
  v += dpp_movf<0xB1>(v);
  v += dpp_movf<0x4E>(v);
  v += dpp_movf<0x141>(v);
  v += dpp_movf<0x140>(v);
  return v;
}

// Async global -> LDS copy (16B per lane), tracked on ASYNCcnt.
__device__ inline void async_copy_b128(const void* gptr, void* lds_ptr) {
  unsigned int ldst = (unsigned int)(unsigned long long)lds_ptr;  // low 32 = LDS addr
  asm volatile("global_load_async_to_lds_b128 %0, %1, off"
               :: "v"(ldst), "v"((unsigned long long)gptr) : "memory");
}

// ---- Tensor Data Mover: one-instruction 2D tile DMA (global -> LDS) ----
// tile_d1 rows of tile_d0 elements (2B each), row stride stride0 elements.
// LDS rows padded: pad_amount(+1) dwords after every 2^(pad_interval+1) dwords.
// tensor dims == tile dims (tiles always fully in-bounds here).
__device__ inline void tdm_load_2d(void* lds, const void* gaddr,
                                   unsigned tile_d0, unsigned tile_d1,
                                   unsigned long long stride0,
                                   unsigned pad_interval, unsigned pad_amount) {
  const unsigned long long ga = (unsigned long long)gaddr;
  u32x4 g0;
  g0[0] = 1u;                                           // count=1 (valid)
  g0[1] = (unsigned)(unsigned long long)lds;            // lds_addr
  g0[2] = (unsigned)ga;                                 // global_addr[31:0]
  g0[3] = (unsigned)((ga >> 32) & 0x1FFFFFFu) | (2u << 30);  // addr[56:32]|type=2
  i32x8 g1;
  g1[0] = (int)((1u << 16)                              // data_size = 2B
              | (1u << 20)                              // pad_enable
              | (pad_interval << 22) | (pad_amount << 25));
  g1[1] = (int)((tile_d0 & 0xFFFFu) << 16);             // tensor_dim0 lo16
  g1[2] = (int)(((tile_d0 >> 16) & 0xFFFFu)             // tensor_dim0 hi16
              | ((tile_d1 & 0xFFFFu) << 16));           // tensor_dim1 lo16
  g1[3] = (int)(((tile_d1 >> 16) & 0xFFFFu)             // tensor_dim1 hi16
              | (tile_d0 << 16));                       // tile_dim0
  g1[4] = (int)(tile_d1 & 0xFFFFu);                     // tile_dim1 (tile_dim2=0)
  g1[5] = (int)(unsigned)stride0;                       // tensor_dim0_stride lo32
  g1[6] = (int)((stride0 >> 32) & 0xFFFFu);             // stride hi16 (dim1_stride=0)
  g1[7] = 0;
  i32x4 gz = {0, 0, 0, 0};
  i32x8 gz8 = {0, 0, 0, 0, 0, 0, 0, 0};
  __builtin_amdgcn_tensor_load_to_lds(g0, g1, gz, gz, gz8, 0);
}

// ---------------- GroupNorm -> f16 ----------------
__global__ __launch_bounds__(256)
void gn_kernel(const float* __restrict__ x, const float* __restrict__ w,
               const float* __restrict__ bb, _Float16* __restrict__ out,
               int chPerGroup, int len) {
  const int groups = 32;
  const int g = blockIdx.x % groups;
  const int b = blockIdx.x / groups;
  const int tid = threadIdx.x;
  const size_t base = ((size_t)b * groups + g) * (size_t)chPerGroup * len;
  const int elems = chPerGroup * len;
  const int n4 = elems >> 2;
  const float4* xv = (const float4*)(x + base);

  float s = 0.f, s2 = 0.f;
  for (int i = tid; i < n4; i += 256) {
    float4 v = xv[i];
    s  += v.x + v.y + v.z + v.w;
    s2 += v.x * v.x + v.y * v.y + v.z * v.z + v.w * v.w;
  }
#pragma unroll
  for (int off = 16; off > 0; off >>= 1) {
    s  += __shfl_xor(s, off, 32);
    s2 += __shfl_xor(s2, off, 32);
  }
  __shared__ float aS[8], aS2[8];
  const int wid = tid >> 5;
  if ((tid & 31) == 0) { aS[wid] = s; aS2[wid] = s2; }
  __syncthreads();
  float ts = 0.f, ts2 = 0.f;
#pragma unroll
  for (int i = 0; i < 8; ++i) { ts += aS[i]; ts2 += aS2[i]; }
  const float mean = ts / (float)elems;
  const float var  = ts2 / (float)elems - mean * mean;
  const float rstd = rsqrtf(var + 1e-5f);

  _Float16* op = out + base;
  for (int i = tid; i < n4; i += 256) {
    float4 v = xv[i];
    const int c = g * chPerGroup + ((i << 2) / len);
    const float wc = w[c] * rstd;
    const float bc = bb[c] - mean * wc;
    v4h hv;
    hv[0] = (_Float16)(v.x * wc + bc);
    hv[1] = (_Float16)(v.y * wc + bc);
    hv[2] = (_Float16)(v.z * wc + bc);
    hv[3] = (_Float16)(v.w * wc + bc);
    *(v4h*)(op + (i << 2)) = hv;
  }
}

// ---------------- WMMA GEMM: acc[m,n] = W[m,:]·X[b,:,n] ----------------
// MODE 0: f16 out [b][M][N], scaled.
// MODE 1: f32 out = resid + gate*(acc+bias), [b][M][N].
// MODE 2: f16 out head-transposed [b][h=m/64][n][d=m%64], scaled (vector stores).
#define BM 64
#define BN 128
#define APITCH 40
#define BPITCH 40

__device__ inline void gemm_stage(const float* __restrict__ W,
                                  const _Float16* __restrict__ X, size_t xbase,
                                  _Float16* smA, _Float16* smB,
                                  int m0, int n0, int k0, int K, int N, int tid) {
  // A: 64x32 f32 -> f16
#pragma unroll
  for (int i = 0; i < 8; ++i) {
    const int idx = tid + i * 256;
    const int r = idx >> 5, kk = idx & 31;
    smA[r * APITCH + kk] = (_Float16)W[(size_t)(m0 + r) * K + (k0 + kk)];
  }
  // B transposed: rows k, cols n -> smB[n][k]
#pragma unroll
  for (int j = 0; j < 2; ++j) {
    const int idx = tid + j * 256;
    const int kk = idx >> 4;
    const int nc = (idx & 15) << 3;
    const _Float16* src = X + xbase + (size_t)(k0 + kk) * N + (n0 + nc);
    if (k0 + 32 < K) __builtin_prefetch(src + (size_t)32 * N, 0, 1);
    v8h v = *(const v8h*)src;
#pragma unroll
    for (int e = 0; e < 8; ++e) smB[(nc + e) * BPITCH + kk] = v[e];
  }
}

template <int MODE>
__global__ __launch_bounds__(256)
void gemm_wmma_kernel(const float* __restrict__ W, const _Float16* __restrict__ X,
                      const float* __restrict__ bias, _Float16* __restrict__ outH,
                      float* __restrict__ outF, const float* __restrict__ resid,
                      const float* __restrict__ gate, float scale,
                      int M, int K, int N) {
  __shared__ __align__(16) _Float16 smA[2][BM * APITCH];
  __shared__ __align__(16) _Float16 smB[2][BN * BPITCH];

  const int tid = threadIdx.x;
  const int lane = tid & 31;
  const int wid = tid >> 5;
  const int wm = wid >> 2;       // 0..1
  const int wn = wid & 3;        // 0..3
  const int hi = lane >> 4;
  const int ln = lane & 15;

  const int n0 = blockIdx.x * BN;
  const int m0 = blockIdx.y * BM;
  const int b  = blockIdx.z;
  const size_t xbase = (size_t)b * K * N;

  v8f acc[2][2];
#pragma unroll
  for (int i = 0; i < 2; ++i)
#pragma unroll
    for (int j = 0; j < 2; ++j) acc[i][j] = v8f_zero();

  const int KT = K >> 5;
  gemm_stage(W, X, xbase, smA[0], smB[0], m0, n0, 0, K, N, tid);

  for (int kt = 0; kt < KT; ++kt) {
    __syncthreads();
    if (kt + 1 < KT)  // overlap next tile's staging with this tile's WMMAs
      gemm_stage(W, X, xbase, smA[(kt + 1) & 1], smB[(kt + 1) & 1],
                 m0, n0, (kt + 1) << 5, K, N, tid);
    const _Float16* cA = smA[kt & 1];
    const _Float16* cB = smB[kt & 1];

    v16h af[2], bf[2];
#pragma unroll
    for (int mt = 0; mt < 2; ++mt)
      af[mt] = frag_a(cA + (wm * 32 + mt * 16 + ln) * APITCH, hi);
#pragma unroll
    for (int nt = 0; nt < 2; ++nt)
      bf[nt] = frag_b(cB + (wn * 32 + nt * 16 + ln) * BPITCH + hi * 16);
#pragma unroll
    for (int mt = 0; mt < 2; ++mt)
#pragma unroll
      for (int nt = 0; nt < 2; ++nt)
        acc[mt][nt] = wmma_f16(af[mt], bf[nt], acc[mt][nt]);
  }

#pragma unroll
  for (int mt = 0; mt < 2; ++mt) {
#pragma unroll
    for (int nt = 0; nt < 2; ++nt) {
      const int ncol = n0 + wn * 32 + nt * 16 + ln;
      if (MODE == 2) {
        // rows r = consecutive d -> one 16B store per tile
        const int hidx = m0 >> 6;
        const int dbase = wm * 32 + mt * 16 + hi * 8;
        v8h pk;
#pragma unroll
        for (int r = 0; r < 8; ++r)
          pk[r] = (_Float16)((acc[mt][nt][r] + bias[m0 + dbase + r]) * scale);
        *(v8h*)(outH + (((size_t)b * NHEAD + hidx) * N + ncol) * HD + dbase) = pk;
      } else {
#pragma unroll
        for (int r = 0; r < 8; ++r) {
          const int mrow = m0 + wm * 32 + mt * 16 + r + hi * 8;
          const float v = acc[mt][nt][r] + bias[mrow];
          const size_t oi = (size_t)b * M * N + (size_t)mrow * N + ncol;
          if (MODE == 0) outH[oi] = (_Float16)(v * scale);
          else           outF[oi] = resid[oi] + gate[b] * v;
        }
      }
    }
  }
}

// ---------------- Flash cross-attention ----------------
// Qt [b][h][q][d] f16 (pre-scaled by 1/sqrt(d)), Kt [b][h][key][d] f16,
// Vb [b][c][key] f16, O [b][c][q] f16.
// Q staged via per-lane async DMA; K/V tiles via single TDM descriptors.
__global__ __launch_bounds__(128)
void attn_kernel(const _Float16* __restrict__ Qt, const _Float16* __restrict__ Kt,
                 const _Float16* __restrict__ Vb, _Float16* __restrict__ O) {
  __shared__ __align__(16) _Float16 sQ[64 * 72];     // [query][d], 144B pitch
  __shared__ __align__(16) _Float16 sK[32 * 72];     // [key][d],   144B pitch
  __shared__ __align__(16) _Float16 sVt[64 * 40];    // [d][key],    80B pitch
  __shared__ __align__(16) _Float16 sP[4 * 16 * 40]; // per-wave P [q][key]

  const int tid = threadIdx.x;
  const int lane = tid & 31, wv = tid >> 5;
  const int hi = lane >> 4, ln = lane & 15;
  const int q0 = blockIdx.x * 64;
  const int h  = blockIdx.y;
  const int b  = blockIdx.z;
  const size_t qtbase = ((size_t)(b * NHEAD + h) * LPIX + q0) * HD;
  const size_t ktbase0 = ((size_t)(b * NHEAD + h) * LCTX) * HD;
  const size_t vhead = ((size_t)b * CCH + h * HD) * LCTX;

  // stage Q tile (64 q x d 0..31 only) straight via async DMA
#pragma unroll
  for (int j = 0; j < 2; ++j) {
    const int idx = tid + j * 128;        // 0..255
    const int q = idx >> 2;
    const int dc = (idx & 3) << 3;        // 16B chunks along d
    async_copy_b128(Qt + qtbase + (size_t)q * HD + dc, &sQ[q * 72 + dc]);
  }
  asm volatile("s_wait_asynccnt 0x0" ::: "memory");
  __syncthreads();

  v16h aq[2];
#pragma unroll
  for (int dc = 0; dc < 2; ++dc)
    aq[dc] = frag_a(sQ + (wv * 16 + ln) * 72 + dc * 32, hi);

  v8f o[4];
  float m[8], l[8];
#pragma unroll
  for (int nt = 0; nt < 4; ++nt) o[nt] = v8f_zero();
#pragma unroll
  for (int r = 0; r < 8; ++r) { m[r] = -1e30f; l[r] = 0.f; }

  for (int kb = 0; kb < LCTX; kb += 32) {
    __syncthreads();
    // K tile: 32 rows x 64 elems (128B rows -> pad 32dw+4dw = 144B pitch)
    // V tile: 64 rows x 32 elems ( 64B rows -> pad 16dw+4dw =  80B pitch)
    // One TDM descriptor each, issued by wave 0 only; TENSORcnt ordered.
    if (wv == 0) {
      tdm_load_2d(&sK[0], Kt + ktbase0 + (size_t)kb * HD,
                  /*tile_d0=*/HD, /*tile_d1=*/32, /*stride0=*/HD,
                  /*pad_interval=*/4, /*pad_amount=*/3);
      tdm_load_2d(&sVt[0], Vb + vhead + kb,
                  /*tile_d0=*/32, /*tile_d1=*/HD, /*stride0=*/LCTX,
                  /*pad_interval=*/3, /*pad_amount=*/3);
      __builtin_amdgcn_s_wait_tensorcnt(0);
    }
    __syncthreads();

    // S tiles (2 x 16 keys), contraction over d = 64 (two K=32 WMMAs)
    v8f s0 = v8f_zero(), s1 = v8f_zero();
#pragma unroll
    for (int dc = 0; dc < 2; ++dc) {
      v16h bk0 = frag_b(sK + (ln)      * 72 + dc * 32 + hi * 16);
      v16h bk1 = frag_b(sK + (16 + ln) * 72 + dc * 32 + hi * 16);
      s0 = wmma_f16(aq[dc], bk0, s0);
      s1 = wmma_f16(aq[dc], bk1, s1);
    }

    // online softmax: rows in accumulator regs, cols across 16-lane half (DPP)
    float cf[8];
#pragma unroll
    for (int r = 0; r < 8; ++r) {
      const float v = row16_max(fmaxf(s0[r], s1[r]));
      const float mn = fmaxf(m[r], v);
      cf[r] = __expf(m[r] - mn);
      m[r] = mn;
    }
#pragma unroll
    for (int nt = 0; nt < 4; ++nt)
#pragma unroll
      for (int r = 0; r < 8; ++r) o[nt][r] *= cf[r];
#pragma unroll
    for (int r = 0; r < 8; ++r) {
      const float p0 = __expf(s0[r] - m[r]);
      const float p1 = __expf(s1[r] - m[r]);
      s0[r] = p0; s1[r] = p1;
      l[r] = l[r] * cf[r] + row16_sum(p0 + p1);
    }

    // D-layout -> A-layout for P via per-wave LDS round trip
    _Float16* pw = sP + wv * 16 * 40;
#pragma unroll
    for (int r = 0; r < 8; ++r) {
      pw[(r + 8 * hi) * 40 + ln]      = (_Float16)s0[r];
      pw[(r + 8 * hi) * 40 + 16 + ln] = (_Float16)s1[r];
    }
    asm volatile("s_wait_dscnt 0x0" ::: "memory");
    v16h ap = frag_a(pw + ln * 40, hi);

    // O += P (16x32) x V (32x64): contraction over 32 keys
#pragma unroll
    for (int nt = 0; nt < 4; ++nt) {
      v16h bv = frag_b(sVt + (nt * 16 + ln) * 40 + hi * 16);
      o[nt] = wmma_f16(ap, bv, o[nt]);
    }
  }

  // finalize: attn_out channel c = h*64 + d, pixel = query.
  float inv[8];
#pragma unroll
  for (int r = 0; r < 8; ++r) inv[r] = 1.0f / l[r];
#pragma unroll
  for (int r = 0; r < 8; r += 2) {
    const int qq = q0 + wv * 16 + r + 8 * hi;
#pragma unroll
    for (int nt = 0; nt < 4; ++nt) {
      const int c = h * HD + nt * 16 + ln;
      v2h pk;
      pk[0] = (_Float16)(o[nt][r]     * inv[r]);
      pk[1] = (_Float16)(o[nt][r + 1] * inv[r + 1]);
      *(v2h*)(O + ((size_t)b * CCH + c) * LPIX + qq) = pk;
    }
  }
}

// ---------------- workspace layout ----------------
#define XN_OFF   ((size_t)0)
#define CXN_OFF  (XN_OFF  + (size_t)BATCH*CCH*LPIX*2)
#define QB_OFF   (CXN_OFF + (size_t)BATCH*CCC*LCTX*2)
#define KB_OFF   (QB_OFF  + (size_t)BATCH*CCH*LPIX*2)
#define VB_OFF   (KB_OFF  + (size_t)BATCH*CCH*LCTX*2)
#define AB_OFF   (VB_OFF  + (size_t)BATCH*CCH*LCTX*2)

extern "C" void kernel_launch(void* const* d_in, const int* in_sizes, int n_in,
                              void* d_out, int out_size, void* d_ws, size_t ws_size,
                              hipStream_t stream) {
  (void)in_sizes; (void)n_in; (void)out_size; (void)ws_size;
  const float* x      = (const float*)d_in[0];
  const float* ctx    = (const float*)d_in[1];
  const float* gate   = (const float*)d_in[2];
  const float* norm_w = (const float*)d_in[3];
  const float* norm_b = (const float*)d_in[4];
  const float* cnw    = (const float*)d_in[5];
  const float* cnb    = (const float*)d_in[6];
  const float* q_w    = (const float*)d_in[7];
  const float* q_b    = (const float*)d_in[8];
  const float* k_w    = (const float*)d_in[9];
  const float* k_b    = (const float*)d_in[10];
  const float* v_w    = (const float*)d_in[11];
  const float* v_b    = (const float*)d_in[12];
  const float* o_w    = (const float*)d_in[13];
  const float* o_b    = (const float*)d_in[14];
  float* out = (float*)d_out;

  char* ws = (char*)d_ws;
  _Float16* xn  = (_Float16*)(ws + XN_OFF);
  _Float16* cxn = (_Float16*)(ws + CXN_OFF);
  _Float16* qb  = (_Float16*)(ws + QB_OFF);   // head-transposed [b][h][q][d]
  _Float16* kb  = (_Float16*)(ws + KB_OFF);   // head-transposed [b][h][key][d]
  _Float16* vb  = (_Float16*)(ws + VB_OFF);   // [b][c][key]
  _Float16* ab  = (_Float16*)(ws + AB_OFF);   // [b][c][q]

  // 1) GroupNorms -> f16
  gn_kernel<<<dim3(BATCH * 32), 256, 0, stream>>>(x,   norm_w, norm_b, xn, 16, LPIX);
  gn_kernel<<<dim3(BATCH * 32), 256, 0, stream>>>(ctx, cnw,    cnb,    cxn, 2, LCTX);

  // 2) Projections: Q/K head-transposed (MODE 2), V row-major (MODE 0).
  //    Q pre-scaled by 1/sqrt(64) = 0.125.
  gemm_wmma_kernel<2><<<dim3(LPIX / BN, CCH / BM, BATCH), 256, 0, stream>>>(
      q_w, xn, q_b, qb, nullptr, nullptr, nullptr, 0.125f, CCH, CCH, LPIX);
  gemm_wmma_kernel<2><<<dim3(LCTX / BN, CCH / BM, BATCH), 256, 0, stream>>>(
      k_w, cxn, k_b, kb, nullptr, nullptr, nullptr, 1.0f, CCH, CCC, LCTX);
  gemm_wmma_kernel<0><<<dim3(LCTX / BN, CCH / BM, BATCH), 256, 0, stream>>>(
      v_w, cxn, v_b, vb, nullptr, nullptr, nullptr, 1.0f, CCH, CCC, LCTX);

  // 3) Flash cross-attention (Q async-DMA, K/V via TDM descriptors)
  attn_kernel<<<dim3(LPIX / 64, NHEAD, BATCH), 128, 0, stream>>>(qb, kb, vb, ab);

  // 4) Out projection + gate + residual (f32 output)
  gemm_wmma_kernel<1><<<dim3(LPIX / BN, CCH / BM, BATCH), 256, 0, stream>>>(
      o_w, ab, o_b, nullptr, out, x, gate, 1.0f, CCH, CCH, LPIX);
}
